// Attention_46248207844114
// MI455X (gfx1250) — compile-verified, matrix operand order checked by reading, not run on hardware
//
#include <hip/hip_runtime.h>
#include <hip/hip_bf16.h>

// ---------------------------------------------------------------------------
// Attention block for MI455X (gfx1250), wave32, WMMA bf16 path.
// B=2, N=2048, D=2048, H=16, HD=128.
// GEMMs: double-buffered LDS pipeline, fragments load as 2 x B128.
// ---------------------------------------------------------------------------

#define Bb   2
#define Nn   2048
#define Dd   2048
#define Hh   16
#define HDd  128
#define QKVD 6144                          // 3*D
#define Mrows 4096                         // B*N
#define KV_ELEMS ((size_t)Bb*Hh*Nn*HDd)    // 8,388,608 elems per cache slab

#define A_TILE (64 * 32)                   // u16 elems per A buffer
#define B_TILE (128 * 16)                  // u32 elems per B buffer

typedef __attribute__((ext_vector_type(16))) __bf16 v16bf;
typedef __attribute__((ext_vector_type(8)))  float  v8f;
typedef __attribute__((ext_vector_type(4)))  unsigned v4u;

union Frag16 { v16bf v; unsigned u[8]; v4u q2[2]; };
union FragC  { v8f v; float f[8]; };

__device__ __forceinline__ unsigned short f2bf(float f) {
  unsigned u = __float_as_uint(f);
  unsigned r = 0x7FFFu + ((u >> 16) & 1u);     // round-to-nearest-even
  return (unsigned short)((u + r) >> 16);
}

__device__ __forceinline__ v8f wmma_bf16(v16bf a, v16bf b, v8f c) {
  // (neg_a, A, neg_b, B, c_mod, C, reuse_a, reuse_b)
  return __builtin_amdgcn_wmma_f32_16x16x32_bf16(false, a, false, b,
                                                 (short)0, c, false, false);
}

// ---------------------------------------------------------------------------
// Stage one 64x32 A tile (row-major bf16) and one 32x128 B tile (column-major
// packed bf16 pairs: Bs32[col*16 + k/2] = {B[k][col], B[k+1][col]}) into LDS.
// ---------------------------------------------------------------------------
__device__ __forceinline__ void stage_tiles(
    const float* __restrict__ A, const float* __restrict__ Bg,
    int K, int Ntot, int m0, int n0, int kt,
    unsigned short* As, unsigned* Bs32)
{
  const int tid = threadIdx.x;
  { // A tile: 8 elems/thread, coalesced fp32 read -> bf16 LDS
    const int row  = tid >> 2;
    const int col0 = (tid & 3) * 8;
    const float* src = A + (size_t)(m0 + row) * K + kt + col0;
    __builtin_prefetch(src + 64, 0, 0);
    unsigned short* dst = As + row * 32 + col0;
#pragma unroll
    for (int i = 0; i < 8; ++i) dst[i] = f2bf(src[i]);
  }
  { // B tile: 2 k-rows x 8 cols per thread, pack pairs in registers
    const int r0 = (tid & 15) * 2;        // k-row pair base (0..30)
    const int c0 = (tid >> 4) * 8;        // 8 columns      (0..120)
    const float* s0 = Bg + (size_t)(kt + r0) * Ntot + n0 + c0;
    const float* s1 = s0 + Ntot;
    __builtin_prefetch(s0 + (size_t)64 * Ntot, 0, 0);
    unsigned* dst = Bs32 + (r0 >> 1);
#pragma unroll
    for (int i = 0; i < 8; ++i) {
      unsigned lo = f2bf(s0[i]);
      unsigned hi = f2bf(s1[i]);
      dst[(c0 + i) * 16] = lo | (hi << 16);
    }
  }
}

// ---------------------------------------------------------------------------
// WMMA GEMM mainloop, double-buffered: C(64x128 per block) = A * B.
// 8 waves: wy = wid>>1 (16-row strip), wx = wid&1 (64-col strip), 4 acc/wave.
// Next K-tile is staged into the alternate LDS buffer while WMMAs consume the
// current one; a single split-barrier per iteration covers both hazards.
// ---------------------------------------------------------------------------
__device__ __forceinline__ void gemm_mainloop(
    const float* __restrict__ A, const float* __restrict__ Bg,
    int K, int Ntot, int m0, int n0,
    unsigned short* As /*2 buffers*/, unsigned* Bs32 /*2 buffers*/,
    FragC acc[4])
{
  const int tid  = threadIdx.x;
  const int lane = tid & 31;
  const int wid  = tid >> 5;
  const int wy   = wid >> 1;
  const int wx   = wid & 1;
  const int r    = lane & 15;
  const int koff = (lane & 16) ? 8 : 0;

#pragma unroll
  for (int t = 0; t < 4; ++t)
#pragma unroll
    for (int j = 0; j < 8; ++j) acc[t].f[j] = 0.f;

  stage_tiles(A, Bg, K, Ntot, m0, n0, 0, As, Bs32);
  __syncthreads();

  for (int kt = 0; kt < K; kt += 32) {
    const int buf = (kt >> 5) & 1;
    const unsigned short* Ac = As + buf * A_TILE;
    const unsigned* Bc = Bs32 + buf * B_TILE;

    if (kt + 32 < K)   // stage next tile into the alternate buffer
      stage_tiles(A, Bg, K, Ntot, m0, n0, kt + 32,
                  As + (buf ^ 1) * A_TILE, Bs32 + (buf ^ 1) * B_TILE);

    // A fragment: lane holds row r; two contiguous 16B chunks
    Frag16 af;
    const unsigned short* arow = Ac + (wy * 16 + r) * 32;
    af.q2[0] = *(const v4u*)(arow + koff);
    af.q2[1] = *(const v4u*)(arow + 16 + koff);

#pragma unroll
    for (int t = 0; t < 4; ++t) {
      Frag16 bf_;
      const int c = wx * 64 + 16 * t + r;          // lane holds column c
      const unsigned* bcol = Bc + c * 16 + (koff >> 1);
      bf_.q2[0] = *(const v4u*)(bcol);
      bf_.q2[1] = *(const v4u*)(bcol + 8);
      acc[t].v = wmma_bf16(af.v, bf_.v, acc[t].v);
    }
    __syncthreads();
  }
}

// ---------------------------------------------------------------------------
// Kernel 1: QKV = x @ Wqkv + bqkv, fused RoPE, scatter to bf16 q/k (row-major
// [b,h,n,hd]) and v transposed ([b,h,hd,n]), plus fp32 pre-RoPE kv cache.
// ---------------------------------------------------------------------------
__global__ void __launch_bounds__(256) qkv_rope_kernel(
    const float* __restrict__ x, const float* __restrict__ Wqkv,
    const float* __restrict__ bqkv, const float* __restrict__ freq_q,
    const float* __restrict__ freq_k, float* __restrict__ kv_out,
    unsigned short* __restrict__ q_bf, unsigned short* __restrict__ k_bf,
    unsigned short* __restrict__ v_t)
{
  __shared__ unsigned short As[2 * A_TILE];
  __shared__ unsigned Bs32[2 * B_TILE];
  FragC acc[4];
  const int m0 = blockIdx.y * 64;
  const int n0 = blockIdx.x * 128;
  gemm_mainloop(x, Wqkv, Dd, QKVD, m0, n0, As, Bs32, acc);

  const int lane = threadIdx.x & 31;
  const int wid  = threadIdx.x >> 5;
  const int wy = wid >> 1, wx = wid & 1;
  const int half = (lane & 16) ? 8 : 0;

#pragma unroll
  for (int t = 0; t < 4; ++t) {
#pragma unroll
    for (int j = 0; j < 8; ++j) {
      const int row = m0 + wy * 16 + j + half;      // token index in [0, B*N)
      const int col = n0 + wx * 64 + 16 * t + (lane & 15);
      float val = acc[t].f[j] + bqkv[col];
      // RoPE pairs (2i,2i+1) differ only in (lane&15)^1 -> wave32 shuffle
      float partner = __shfl_xor(val, 1, 32);
      const int b = row >> 11, n = row & 2047;
      const int sec = col >> 11;                     // 0=q 1=k 2=v
      const int cc = col & 2047;
      const int h = cc >> 7, hd = cc & 127;
      const size_t idx = (((size_t)(b * Hh + h)) * Nn + n) * HDd + hd;
      if (sec == 2) {
        kv_out[KV_ELEMS + idx] = val;               // v cache (fp32)
        v_t[(((size_t)(b * Hh + h)) * HDd + hd) * Nn + n] = f2bf(val);
      } else {
        const float* fr = (sec == 0) ? freq_q : freq_k;
        float ang = fr[n * (HDd / 2) + (hd >> 1)];
        float cs = __cosf(ang), sn = __sinf(ang);
        float roped = (hd & 1) ? (partner * sn + val * cs)
                               : (val * cs - partner * sn);
        if (sec == 0) {
          q_bf[idx] = f2bf(roped);
        } else {
          kv_out[idx] = val;                        // pre-RoPE k cache
          k_bf[idx] = f2bf(roped);
        }
      }
    }
  }
}

// ---------------------------------------------------------------------------
// Kernel 2: flash attention. One wave per 16-query tile, 32-key chunks,
// online softmax, P transposed through per-wave LDS, O accumulated in fp32.
// ---------------------------------------------------------------------------
__global__ void __launch_bounds__(256) flash_attn_kernel(
    const unsigned short* __restrict__ q_bf,   // [b,h,n,hd]
    const unsigned short* __restrict__ k_bf,   // [b,h,n,hd]  (roped)
    const unsigned short* __restrict__ v_t,    // [b,h,hd,n]  (transposed)
    const int* __restrict__ attn_mask,         // [B,1,1,N]
    float* __restrict__ attn_o)                // [B,N,D]
{
  __shared__ unsigned short Pt[8 * 16 * 32];   // 1KB per wave P tile
  const int lane = threadIdx.x & 31;
  const int wid  = threadIdx.x >> 5;
  const int b = blockIdx.z, h = blockIdx.y;
  const int q0 = blockIdx.x * 128 + wid * 16;
  const int r = lane & 15;
  const int koff = (lane & 16) ? 8 : 0;
  const int half = (lane & 16) ? 8 : 0;
  const size_t bh = (size_t)(b * Hh + h);

  const unsigned short* qbase = q_bf + (bh * Nn + q0) * HDd;
  const unsigned short* kbase = k_bf + bh * Nn * HDd;
  const unsigned short* vbase = v_t + bh * HDd * Nn;

  // 4 Q A-fragments covering HD=128 in chunks of 32 (2 x B128 each)
  Frag16 qa[4];
  const unsigned short* qrow = qbase + (size_t)r * HDd;
#pragma unroll
  for (int c = 0; c < 4; ++c) {
    qa[c].q2[0] = *(const v4u*)(qrow + 32 * c + koff);
    qa[c].q2[1] = *(const v4u*)(qrow + 32 * c + 16 + koff);
  }

  FragC O[8];
  float mrow[8], lrow[8];
#pragma unroll
  for (int j = 0; j < 8; ++j) { mrow[j] = -3.402823466e38f; lrow[j] = 0.f; }
#pragma unroll
  for (int f = 0; f < 8; ++f)
#pragma unroll
    for (int j = 0; j < 8; ++j) O[f].f[j] = 0.f;

  const float inv_sqrt_hd = 0.08838834764831845f;
  const int nkt = (q0 + 15) / 32 + 1;             // causal key chunks

  for (int kt = 0; kt < nkt; ++kt) {
    const int key0 = kt * 32;
    // ---- S = Q @ K^T for 2 key sub-tiles of 16 ----
    FragC S[2];
#pragma unroll
    for (int s = 0; s < 2; ++s) {
#pragma unroll
      for (int j = 0; j < 8; ++j) S[s].f[j] = 0.f;
      const int kkcol = key0 + 16 * s + r;        // lane's key column
      const unsigned short* krow = kbase + (size_t)kkcol * HDd;
#pragma unroll
      for (int c = 0; c < 4; ++c) {
        Frag16 kf;
        kf.q2[0] = *(const v4u*)(krow + 32 * c + koff);
        kf.q2[1] = *(const v4u*)(krow + 32 * c + 16 + koff);
        S[s].v = wmma_bf16(qa[c].v, kf.v, S[s].v);
      }
    }
    // ---- mask + online softmax (row = j+half, col = lane&15) ----
#pragma unroll
    for (int j = 0; j < 8; ++j) {
      const int q = q0 + j + half;
      const int k0c = key0 + r;
      const int k1c = key0 + 16 + r;
      const bool v0 = (k0c <= q) && (attn_mask[b * Nn + k0c] != 0);
      const bool v1 = (k1c <= q) && (attn_mask[b * Nn + k1c] != 0);
      float a0 = v0 ? S[0].f[j] * inv_sqrt_hd : -3.402823466e38f;
      float a1 = v1 ? S[1].f[j] * inv_sqrt_hd : -3.402823466e38f;
      float t = fmaxf(a0, a1);
#pragma unroll
      for (int mm = 1; mm <= 8; mm <<= 1) t = fmaxf(t, __shfl_xor(t, mm, 32));
      const float mnew = fmaxf(mrow[j], t);
      const float sc = __expf(mrow[j] - mnew);
      const float p0 = v0 ? __expf(a0 - mnew) : 0.f;
      const float p1 = v1 ? __expf(a1 - mnew) : 0.f;
      float rs = p0 + p1;
#pragma unroll
      for (int mm = 1; mm <= 8; mm <<= 1) rs += __shfl_xor(rs, mm, 32);
      lrow[j] = lrow[j] * sc + rs;
      mrow[j] = mnew;
#pragma unroll
      for (int f = 0; f < 8; ++f) O[f].f[j] *= sc;
      const int rr = j + half;
      Pt[wid * 512 + rr * 32 + r]      = f2bf(p0);
      Pt[wid * 512 + rr * 32 + 16 + r] = f2bf(p1);
    }
    // ---- P (16x32) as A-fragment via per-wave LDS transpose (2 x B128) ----
    Frag16 pf;
    {
      const unsigned short* prow = &Pt[wid * 512 + r * 32];
      pf.q2[0] = *(const v4u*)(prow + koff);
      pf.q2[1] = *(const v4u*)(prow + 16 + koff);
    }
    // ---- O += P @ V : V fragment from transposed layout (2 x B128) ----
#pragma unroll
    for (int f = 0; f < 8; ++f) {
      Frag16 vf;
      const int dc = 16 * f + r;                  // lane's d column
      const unsigned short* vcol = vbase + (size_t)dc * Nn + key0;
      vf.q2[0] = *(const v4u*)(vcol + koff);
      vf.q2[1] = *(const v4u*)(vcol + 16 + koff);
      O[f].v = wmma_bf16(pf.v, vf.v, O[f].v);
    }
  }

  // ---- normalize + write attn_o[b][q][h*HD + d] (fp32) ----
#pragma unroll
  for (int f = 0; f < 8; ++f)
#pragma unroll
    for (int j = 0; j < 8; ++j) {
      const int q = q0 + j + half;
      const int col = h * HDd + 16 * f + r;
      const float denom = lrow[j];
      const float ov = (denom > 0.f) ? O[f].f[j] / denom : 0.f;
      attn_o[((size_t)b * Nn + q) * Dd + col] = ov;
    }
}

// ---------------------------------------------------------------------------
// Kernel 3: out = attn_o @ Wout + bout
// ---------------------------------------------------------------------------
__global__ void __launch_bounds__(256) out_proj_kernel(
    const float* __restrict__ attn_o, const float* __restrict__ Wout,
    const float* __restrict__ bout, float* __restrict__ out)
{
  __shared__ unsigned short As[2 * A_TILE];
  __shared__ unsigned Bs32[2 * B_TILE];
  FragC acc[4];
  const int m0 = blockIdx.y * 64;
  const int n0 = blockIdx.x * 128;
  gemm_mainloop(attn_o, Wout, Dd, Dd, m0, n0, As, Bs32, acc);

  const int lane = threadIdx.x & 31;
  const int wid  = threadIdx.x >> 5;
  const int wy = wid >> 1, wx = wid & 1;
  const int half = (lane & 16) ? 8 : 0;
#pragma unroll
  for (int t = 0; t < 4; ++t)
#pragma unroll
    for (int j = 0; j < 8; ++j) {
      const int row = m0 + wy * 16 + j + half;
      const int col = n0 + wx * 64 + 16 * t + (lane & 15);
      out[(size_t)row * Dd + col] = acc[t].f[j] + bout[col];
    }
}

// ---------------------------------------------------------------------------
// Launch: d_in order = x, freq_q, freq_k, expanded_attn_masks, casual_mask,
// Wqkv, bqkv, Wout, bout. d_out = [out (B*N*D) | next_prefix_kv (2*B*H*N*HD)].
// ---------------------------------------------------------------------------
extern "C" void kernel_launch(void* const* d_in, const int* in_sizes, int n_in,
                              void* d_out, int out_size, void* d_ws, size_t ws_size,
                              hipStream_t stream) {
  (void)in_sizes; (void)n_in; (void)out_size; (void)ws_size;
  const float* x       = (const float*)d_in[0];
  const float* freq_q  = (const float*)d_in[1];
  const float* freq_k  = (const float*)d_in[2];
  const int*   amask   = (const int*)d_in[3];
  /* d_in[4] = causal mask (tril) — computed analytically in-kernel */
  const float* Wqkv    = (const float*)d_in[5];
  const float* bqkv    = (const float*)d_in[6];
  const float* Wout    = (const float*)d_in[7];
  const float* bout    = (const float*)d_in[8];

  float* out    = (float*)d_out;
  float* kv_out = out + (size_t)Bb * Nn * Dd;      // [2,B,H,N,HD]

  unsigned short* q_bf = (unsigned short*)d_ws;     // bf16 [B,H,N,HD]
  unsigned short* k_bf = q_bf + KV_ELEMS;           // bf16 [B,H,N,HD]
  unsigned short* v_t  = k_bf + KV_ELEMS;           // bf16 [B,H,HD,N]
  float* attn_o = (float*)(v_t + KV_ELEMS);         // fp32 [B,N,D]

  dim3 blk(256);
  qkv_rope_kernel<<<dim3(QKVD / 128, Mrows / 64), blk, 0, stream>>>(
      x, Wqkv, bqkv, freq_q, freq_k, kv_out, q_bf, k_bf, v_t);
  flash_attn_kernel<<<dim3(Nn / 128, Hh, Bb), blk, 0, stream>>>(
      q_bf, k_bf, v_t, amask, attn_o);
  out_proj_kernel<<<dim3(Dd / 128, Mrows / 64), blk, 0, stream>>>(
      attn_o, Wout, bout, out);
}
